// XLMultiHeadAttention_29454885715996
// MI455X (gfx1250) — compile-verified, matrix operand order checked by reading, not run on hardware
//
#include <hip/hip_runtime.h>

// ---------------------------------------------------------------------------
// Transformer-XL relative multi-head attention for MI455X (gfx1250, wave32).
// - bf16 WMMA (v_wmma_f32_16x16x32_bf16) for all GEMM-shaped work, f32 accum.
// - GLOBAL_LOAD_ASYNC_TO_LDS_B128 (ASYNCcnt) + double-buffered LDS ping-pong
//   to overlap HBM latency with WMMA issue.
// - BD tensor materialized [i][j][bn] (bn innermost => 32 f32 = one 128B line
//   shared by all head-batches) for full-line utilization in the rel-shift
//   gather; grid ordered bn-fastest for L2 reuse of indices/lower/BD lines.
// - Flash-style attention: scores never hit HBM.
// ---------------------------------------------------------------------------

typedef __bf16 bf16_t;
typedef __attribute__((ext_vector_type(4)))  __bf16 v4bf;
typedef __attribute__((ext_vector_type(8)))  __bf16 v8bf;
typedef __attribute__((ext_vector_type(16))) __bf16 v16bf;
typedef __attribute__((ext_vector_type(8)))  float  v8f;

#define N_HEADS 16
#define KEY_DIM 64
#define HIDDEN  1024
#define B_SZ    2
#define LQ      2048
#define LK      2048
#define R_SZ    2048
#define BN      32           // B_SZ * N_HEADS
#define XSCALE  0.125f       // 1/sqrt(64)

// --- CDNA5 async copy helpers ----------------------------------------------

// 16B global -> LDS DMA, tracked by ASYNCcnt (ISA 10.7 / 15.18.3 op 98).
// Generic LDS pointer low 32 bits == LDS byte address (aperture rule, ISA 10.2).
__device__ __forceinline__ void async_copy_b128(void* lds_dst, const void* gsrc) {
    uint32_t l = (uint32_t)(uintptr_t)lds_dst;
    asm volatile("global_load_async_to_lds_b128 %0, %1, off"
                 :: "v"(l), "v"(gsrc) : "memory");
}
__device__ __forceinline__ void wait_async0() {
    asm volatile("s_wait_asynccnt 0x0" ::: "memory");
}

// --- WMMA helpers -----------------------------------------------------------

// Assemble a 16x32 bf16 A/B fragment from two contiguous 8-element chunks,
// matching the CDNA5 16-bit operand layout (ISA 7.12.2):
//   element e -> K = (e>>3)*16 + (e&7) + 8*(lane>=16)
__device__ __forceinline__ v16bf make_frag(const bf16_t* p0, const bf16_t* p1) {
    v8bf lo = *reinterpret_cast<const v8bf*>(p0);
    v8bf hi = *reinterpret_cast<const v8bf*>(p1);
    v16bf r;
#pragma unroll
    for (int e = 0; e < 8; ++e) { r[e] = lo[e]; r[e + 8] = hi[e]; }
    return r;
}

__device__ __forceinline__ v8f wmma_bf16(v16bf a, v16bf b, v8f c) {
    return __builtin_amdgcn_wmma_f32_16x16x32_bf16(
        false, a, false, b, (short)0, c, false, false);
}

// Reductions across the 16 lanes of a wave32 half (C-matrix row lives in
// lanes 0..15 or 16..31 of one VGPR).
__device__ __forceinline__ float half_max(float v) {
    v = fmaxf(v, __shfl_xor(v, 1, 32));
    v = fmaxf(v, __shfl_xor(v, 2, 32));
    v = fmaxf(v, __shfl_xor(v, 4, 32));
    v = fmaxf(v, __shfl_xor(v, 8, 32));
    return v;
}
__device__ __forceinline__ float half_sum(float v) {
    v += __shfl_xor(v, 1, 32);
    v += __shfl_xor(v, 2, 32);
    v += __shfl_xor(v, 4, 32);
    v += __shfl_xor(v, 8, 32);
    return v;
}

__device__ __forceinline__ v8f vzero8() {
    return v8f{0.f,0.f,0.f,0.f,0.f,0.f,0.f,0.f};
}

// --- f32 -> bf16 conversion -------------------------------------------------

__global__ __launch_bounds__(256) void cvt_kernel(const float4* __restrict__ src,
                                                  bf16_t* __restrict__ dst, int n4) {
    int i = blockIdx.x * 256 + threadIdx.x;
    if (i < n4) {
        float4 f = src[i];
        v4bf o = { (bf16_t)f.x, (bf16_t)f.y, (bf16_t)f.z, (bf16_t)f.w };
        *reinterpret_cast<v4bf*>(&dst[(size_t)i * 4]) = o;
    }
}

// --- Generic [M,1024] x [1024,1024] bf16 GEMM, 64x64 tile / 4 waves ---------
// MODE 0: dual bf16 outputs with per-column offsets (Q: +u_weight / +v_weight)
// MODE 1: single bf16 output (K / V / R projections)
// MODE 2: f32 output (final @Wo)
// A-tile fills are async-to-LDS; B-tile is global->VGPR->LDS transpose.
// Double-buffered: tile i+1 in flight while WMMAs consume tile i.

template<int MODE>
__global__ __launch_bounds__(128) void gemm1024_bf16_kernel(
    const bf16_t* __restrict__ A, const bf16_t* __restrict__ W,
    const float* __restrict__ bias,
    const float* __restrict__ off0, const float* __restrict__ off1,
    bf16_t* __restrict__ out0, bf16_t* __restrict__ out1,
    float* __restrict__ outf) {
    __shared__ alignas(16) bf16_t As[2][64][32];   // [buf][m][k]
    __shared__ alignas(16) bf16_t Bs[2][64][32];   // [buf][n][k] (transposed)

    const int tid  = threadIdx.x;
    const int wave = tid >> 5, lid = tid & 31;
    const int half = lid >> 4, n15 = lid & 15;
    const int bi = blockIdx.x >> 4;
    const int bj = blockIdx.x & 15;
    const int rowbase = bi * 64, colbase = bj * 64;

    const int ar = tid >> 1, ah = tid & 1;     // A loader: 64 rows x 2 halves
    const int bk_ = tid >> 2, bq = tid & 3;    // B loader: 32 k x 4 quarters

    auto issueA = [&](int kb, int buf) {
        const bf16_t* g = &A[(size_t)(rowbase + ar) * HIDDEN + kb + ah * 16];
        async_copy_b128(&As[buf][ar][ah * 16],     g);
        async_copy_b128(&As[buf][ar][ah * 16 + 8], g + 8);
    };
    auto loadB = [&](int kb, v8bf& w0, v8bf& w1) {
        const bf16_t* s = &W[(size_t)(kb + bk_) * HIDDEN + colbase + bq * 16];
        w0 = *reinterpret_cast<const v8bf*>(s);
        w1 = *reinterpret_cast<const v8bf*>(s + 8);
    };
    auto storeB = [&](int buf, v8bf w0, v8bf w1) {
#pragma unroll
        for (int i = 0; i < 8; ++i) {
            Bs[buf][bq * 16 + i][bk_]     = w0[i];
            Bs[buf][bq * 16 + 8 + i][bk_] = w1[i];
        }
    };

    v8f acc[4];
#pragma unroll
    for (int t = 0; t < 4; ++t) acc[t] = vzero8();

    // Prologue: tile 0
    issueA(0, 0);
    {
        v8bf w0, w1;
        loadB(0, w0, w1);
        storeB(0, w0, w1);
    }
    wait_async0();
    __syncthreads();

    for (int kb = 0; kb < HIDDEN; kb += 32) {
        const int buf = (kb >> 5) & 1;
        const bool more = (kb + 32) < HIDDEN;
        v8bf w0, w1;
        if (more) {
            issueA(kb + 32, buf ^ 1);          // in flight during WMMAs
            loadB(kb + 32, w0, w1);            // latency hidden by compute
        }
        v16bf a = make_frag(&As[buf][wave * 16 + n15][8 * half],
                            &As[buf][wave * 16 + n15][16 + 8 * half]);
#pragma unroll
        for (int nt = 0; nt < 4; ++nt) {
            v16bf b = make_frag(&Bs[buf][nt * 16 + n15][8 * half],
                                &Bs[buf][nt * 16 + n15][16 + 8 * half]);
            acc[nt] = wmma_bf16(a, b, acc[nt]);
        }
        if (more) storeB(buf ^ 1, w0, w1);
        wait_async0();
        __syncthreads();
    }

    // Epilogue: C layout -> VGPR r holds row (r + 8*half), lanes hold columns.
#pragma unroll
    for (int nt = 0; nt < 4; ++nt) {
        int col  = colbase + nt * 16 + n15;
        float bb = bias[col];
        float o0 = (MODE == 0) ? off0[col] : 0.f;
        float o1 = (MODE == 0) ? off1[col] : 0.f;
#pragma unroll
        for (int r = 0; r < 8; ++r) {
            size_t row = rowbase + wave * 16 + r + 8 * half;
            float v = acc[nt][r] + bb;
            if (MODE == 2) {
                outf[row * HIDDEN + col] = v;
            } else if (MODE == 1) {
                out0[row * HIDDEN + col] = (bf16_t)v;
            } else {
                out0[row * HIDDEN + col] = (bf16_t)(v + o0);
                out1[row * HIDDEN + col] = (bf16_t)(v + o1);
            }
        }
    }
}

// --- BD = (q + v_weight) @ r^T, per (b,n): M=2048, N=R=2048, K=64 -----------
// Both operand tiles are straight copies -> fully async-to-LDS, both K-steps
// issued up front, single asynccnt wait, then 8 back-to-back WMMAs.
// Output layout BD[i][jr][bn].

__global__ __launch_bounds__(128) void bd_gemm_kernel(
    const bf16_t* __restrict__ qv, const bf16_t* __restrict__ rb,
    float* __restrict__ BD) {
    __shared__ alignas(16) bf16_t As[2][64][32];   // [kstep][m][k]
    __shared__ alignas(16) bf16_t Bs[2][64][32];   // [kstep][jr][k]

    const int tid  = threadIdx.x;
    const int wave = tid >> 5, lid = tid & 31;
    const int half = lid >> 4, n15 = lid & 15;
    const int bn = blockIdx.x & 31;
    const int ti = (blockIdx.x >> 5) & 31;
    const int tj = blockIdx.x >> 10;
    const int b = bn >> 4, n = bn & 15;
    const int ibase = ti * 64, jbase = tj * 64;

    const int r = tid >> 1, h = tid & 1;
#pragma unroll
    for (int ks = 0; ks < 2; ++ks) {
        const bf16_t* ga =
            &qv[(size_t)(b * LQ + ibase + r) * HIDDEN + n * KEY_DIM + ks * 32 + h * 16];
        async_copy_b128(&As[ks][r][h * 16],     ga);
        async_copy_b128(&As[ks][r][h * 16 + 8], ga + 8);
        const bf16_t* gb =
            &rb[(size_t)(jbase + r) * HIDDEN + n * KEY_DIM + ks * 32 + h * 16];
        async_copy_b128(&Bs[ks][r][h * 16],     gb);
        async_copy_b128(&Bs[ks][r][h * 16 + 8], gb + 8);
    }
    wait_async0();
    __syncthreads();

    v8f acc[4];
#pragma unroll
    for (int t = 0; t < 4; ++t) acc[t] = vzero8();
#pragma unroll
    for (int ks = 0; ks < 2; ++ks) {
        v16bf a = make_frag(&As[ks][wave * 16 + n15][8 * half],
                            &As[ks][wave * 16 + n15][16 + 8 * half]);
#pragma unroll
        for (int nt = 0; nt < 4; ++nt) {
            v16bf bfr = make_frag(&Bs[ks][nt * 16 + n15][8 * half],
                                  &Bs[ks][nt * 16 + n15][16 + 8 * half]);
            acc[nt] = wmma_bf16(a, bfr, acc[nt]);
        }
    }
#pragma unroll
    for (int nt = 0; nt < 4; ++nt) {
        int jr = jbase + nt * 16 + n15;
#pragma unroll
        for (int rr = 0; rr < 8; ++rr) {
            size_t i = ibase + wave * 16 + rr + 8 * half;
            BD[(i * R_SZ + jr) * BN + bn] = acc[nt][rr];
        }
    }
}

// --- Flash attention: AC-WMMA + BD-gather + online softmax + P.V-WMMA ------
// Double-buffered K/V tiles: K-tile async-to-LDS, V-tile via VGPR transpose,
// next tile in flight while the current 32-key step computes.

__global__ __launch_bounds__(128) void attn_kernel(
    const bf16_t* __restrict__ qu, const bf16_t* __restrict__ kbuf,
    const bf16_t* __restrict__ vbuf, const float* __restrict__ BD,
    const int* __restrict__ ind, const float* __restrict__ lower,
    bf16_t* __restrict__ attn_out) {
    __shared__ alignas(16) bf16_t Ks[2][32][64];    // [buf][j_local][kd]
    __shared__ alignas(16) bf16_t Vs[2][64][32];    // [buf][d][j_local]
    __shared__ alignas(16) bf16_t Ps[4][16][32];    // per-wave probs, [m][j]

    const int tid  = threadIdx.x;
    const int wave = tid >> 5, lid = tid & 31;
    const int half = lid >> 4, n15 = lid & 15;
    const int bn = blockIdx.x & 31;
    const int ib = blockIdx.x >> 5;
    const int b = bn >> 4, n = bn & 15;
    const int ibase = ib * 64;
    const int mrow = ibase + wave * 16 + n15;       // lane's A-matrix row

    const int jl = tid >> 2, q = tid & 3;           // tile loader coords

    auto issueK = [&](int jb, int buf) {
        const bf16_t* g =
            &kbuf[(size_t)(b * LK + jb + jl) * HIDDEN + n * KEY_DIM + q * 16];
        async_copy_b128(&Ks[buf][jl][q * 16],     g);
        async_copy_b128(&Ks[buf][jl][q * 16 + 8], g + 8);
    };
    auto loadV = [&](int jb, v8bf& v0, v8bf& v1) {
        const bf16_t* s =
            &vbuf[(size_t)(b * LK + jb + jl) * HIDDEN + n * KEY_DIM + q * 16];
        v0 = *reinterpret_cast<const v8bf*>(s);
        v1 = *reinterpret_cast<const v8bf*>(s + 8);
    };
    auto storeV = [&](int buf, v8bf v0, v8bf v1) {
#pragma unroll
        for (int i = 0; i < 8; ++i) {
            Vs[buf][q * 16 + i][jl]     = v0[i];
            Vs[buf][q * 16 + 8 + i][jl] = v1[i];
        }
    };

    // Q fragments (K = 0..31 and 32..63 of the head), loaded once.
    const bf16_t* qp = &qu[(size_t)(b * LQ + mrow) * HIDDEN + n * KEY_DIM];
    v16bf aq0 = make_frag(qp + 8 * half,      qp + 16 + 8 * half);
    v16bf aq1 = make_frag(qp + 32 + 8 * half, qp + 48 + 8 * half);

    float mst[8], lst[8];
    v8f ov[4];
#pragma unroll
    for (int r = 0; r < 8; ++r) { mst[r] = -3.0e38f; lst[r] = 0.f; }
#pragma unroll
    for (int t = 0; t < 4; ++t) ov[t] = vzero8();

    // Prologue: tile 0
    issueK(0, 0);
    {
        v8bf v0, v1;
        loadV(0, v0, v1);
        storeV(0, v0, v1);
    }
    wait_async0();
    __syncthreads();

    for (int jb0 = 0; jb0 < LK; jb0 += 32) {
        const int buf = (jb0 >> 5) & 1;
        const bool more = (jb0 + 32) < LK;
        v8bf nv0, nv1;
        if (more) {
            issueK(jb0 + 32, buf ^ 1);
            loadV(jb0 + 32, nv0, nv1);
        }

        // AC scores: two 16x16 j-tiles, K=64 (2 WMMA each)
        v8f s0 = vzero8(), s1 = vzero8();
        {
            v16bf bk0 = make_frag(&Ks[buf][n15][8 * half],      &Ks[buf][n15][16 + 8 * half]);
            v16bf bk1 = make_frag(&Ks[buf][n15][32 + 8 * half], &Ks[buf][n15][48 + 8 * half]);
            s0 = wmma_bf16(aq0, bk0, s0);
            s0 = wmma_bf16(aq1, bk1, s0);
            v16bf bk2 = make_frag(&Ks[buf][16 + n15][8 * half],      &Ks[buf][16 + n15][16 + 8 * half]);
            v16bf bk3 = make_frag(&Ks[buf][16 + n15][32 + 8 * half], &Ks[buf][16 + n15][48 + 8 * half]);
            s1 = wmma_bf16(aq0, bk2, s1);
            s1 = wmma_bf16(aq1, bk3, s1);
        }

        // BD gather + mask + scale, then online softmax row update
#pragma unroll
        for (int r = 0; r < 8; ++r) {
            int i = ibase + wave * 16 + r + 8 * half;
            int p0i = i * LK + (jb0 + n15);
            int gi = ind[p0i], gj = ind[LQ * LK + p0i];
            float bd = BD[((size_t)gi * R_SZ + gj) * BN + bn] * lower[p0i];
            s0[r] = (s0[r] + bd) * XSCALE;
            int p1i = i * LK + (jb0 + 16 + n15);
            gi = ind[p1i]; gj = ind[LQ * LK + p1i];
            bd = BD[((size_t)gi * R_SZ + gj) * BN + bn] * lower[p1i];
            s1[r] = (s1[r] + bd) * XSCALE;

            float mt   = half_max(fmaxf(s0[r], s1[r]));
            float mnew = fmaxf(mst[r], mt);
            float alpha = __expf(mst[r] - mnew);
            float p0 = __expf(s0[r] - mnew);
            float p1 = __expf(s1[r] - mnew);
            lst[r] = lst[r] * alpha + half_sum(p0 + p1);
            mst[r] = mnew;
#pragma unroll
            for (int dt = 0; dt < 4; ++dt) ov[dt][r] *= alpha;
            // C layout -> plain [m][j] in per-wave LDS (A-layout round trip)
            Ps[wave][r + 8 * half][n15]      = (bf16_t)p0;
            Ps[wave][r + 8 * half][16 + n15] = (bf16_t)p1;
        }

        // P.V accumulation: A = probs 16x32, B = V^T tile, N over VAL_DIM
        v16bf pa = make_frag(&Ps[wave][n15][8 * half], &Ps[wave][n15][16 + 8 * half]);
#pragma unroll
        for (int dt = 0; dt < 4; ++dt) {
            v16bf bv = make_frag(&Vs[buf][dt * 16 + n15][8 * half],
                                 &Vs[buf][dt * 16 + n15][16 + 8 * half]);
            ov[dt] = wmma_bf16(pa, bv, ov[dt]);
        }

        if (more) storeV(buf ^ 1, nv0, nv1);
        wait_async0();
        __syncthreads();
    }

    // Normalize and emit bf16 head output for the final @Wo GEMM
#pragma unroll
    for (int r = 0; r < 8; ++r) {
        float inv = 1.f / lst[r];
        size_t row = b * LQ + ibase + wave * 16 + r + 8 * half;
#pragma unroll
        for (int dt = 0; dt < 4; ++dt) {
            attn_out[row * HIDDEN + n * KEY_DIM + dt * 16 + n15] =
                (bf16_t)(ov[dt][r] * inv);
        }
    }
}

// ---------------------------------------------------------------------------

extern "C" void kernel_launch(void* const* d_in, const int* in_sizes, int n_in,
                              void* d_out, int out_size, void* d_ws, size_t ws_size,
                              hipStream_t stream) {
    const float* query = (const float*)d_in[0];
    const float* value = (const float*)d_in[1];
    const float* pemb  = (const float*)d_in[2];
    const float* u_w   = (const float*)d_in[3];
    const float* v_w   = (const float*)d_in[4];
    const int*   ind   = (const int*)d_in[5];
    const float* lower = (const float*)d_in[6];
    const float* Wq = (const float*)d_in[7];  const float* bq = (const float*)d_in[8];
    const float* Wk = (const float*)d_in[9];  const float* bk = (const float*)d_in[10];
    const float* Wv = (const float*)d_in[11]; const float* bv = (const float*)d_in[12];
    const float* Wr = (const float*)d_in[13]; const float* br = (const float*)d_in[14];
    const float* Wo = (const float*)d_in[15]; const float* bo = (const float*)d_in[16];
    float* out = (float*)d_out;

    const size_t actN = (size_t)B_SZ * LQ * HIDDEN;   // 4M elems
    const size_t posN = (size_t)R_SZ * HIDDEN;        // 2M
    const size_t wN   = (size_t)HIDDEN * HIDDEN;      // 1M

    char* w = (char*)d_ws;
    float* BD = (float*)w;            w += (size_t)LQ * R_SZ * BN * 4;   // 512 MB
    bf16_t* query_bf = (bf16_t*)w;    w += actN * 2;
    bf16_t* value_bf = (bf16_t*)w;    w += actN * 2;
    bf16_t* pos_bf   = (bf16_t*)w;    w += posN * 2;
    bf16_t* Wq_bf    = (bf16_t*)w;    w += wN * 2;
    bf16_t* Wk_bf    = (bf16_t*)w;    w += wN * 2;
    bf16_t* Wv_bf    = (bf16_t*)w;    w += wN * 2;
    bf16_t* Wr_bf    = (bf16_t*)w;    w += wN * 2;
    bf16_t* Wo_bf    = (bf16_t*)w;    w += wN * 2;
    bf16_t* qu_bf    = (bf16_t*)w;    w += actN * 2;   // q + bias + u
    bf16_t* qv_bf    = (bf16_t*)w;    w += actN * 2;   // q + bias + v
    bf16_t* k_bf     = (bf16_t*)w;    w += actN * 2;
    bf16_t* v_bf     = (bf16_t*)w;    w += actN * 2;
    bf16_t* r_bf     = (bf16_t*)w;    w += posN * 2;
    bf16_t* attn_bf  = (bf16_t*)w;    w += actN * 2;

    auto cvt = [&](const float* src, bf16_t* dst, size_t n) {
        int n4 = (int)(n / 4);
        cvt_kernel<<<dim3((n4 + 255) / 256), dim3(256), 0, stream>>>(
            (const float4*)src, dst, n4);
    };
    cvt(query, query_bf, actN);
    cvt(value, value_bf, actN);
    cvt(pemb,  pos_bf,   posN);
    cvt(Wq, Wq_bf, wN);  cvt(Wk, Wk_bf, wN);  cvt(Wv, Wv_bf, wN);
    cvt(Wr, Wr_bf, wN);  cvt(Wo, Wo_bf, wN);

    // Projections (dual-output Q fuses +u_weight / +v_weight into epilogue)
    gemm1024_bf16_kernel<0><<<dim3((B_SZ * LQ / 64) * 16), dim3(128), 0, stream>>>(
        query_bf, Wq_bf, bq, u_w, v_w, qu_bf, qv_bf, nullptr);
    gemm1024_bf16_kernel<1><<<dim3((B_SZ * LK / 64) * 16), dim3(128), 0, stream>>>(
        value_bf, Wk_bf, bk, nullptr, nullptr, k_bf, nullptr, nullptr);
    gemm1024_bf16_kernel<1><<<dim3((B_SZ * LK / 64) * 16), dim3(128), 0, stream>>>(
        value_bf, Wv_bf, bv, nullptr, nullptr, v_bf, nullptr, nullptr);
    gemm1024_bf16_kernel<1><<<dim3((R_SZ / 64) * 16), dim3(128), 0, stream>>>(
        pos_bf, Wr_bf, br, nullptr, nullptr, r_bf, nullptr, nullptr);

    // BD[i][jr][bn] materialization (gather-friendly layout)
    bd_gemm_kernel<<<dim3(BN * (LQ / 64) * (R_SZ / 64)), dim3(128), 0, stream>>>(
        qv_bf, r_bf, BD);

    // Fused flash attention (AC wmma + BD gather + softmax + P.V wmma)
    attn_kernel<<<dim3(BN * (LQ / 64)), dim3(128), 0, stream>>>(
        qu_bf, k_bf, v_bf, BD, ind, lower, attn_bf);

    // Output projection: f32 result
    gemm1024_bf16_kernel<2><<<dim3((B_SZ * LQ / 64) * 16), dim3(128), 0, stream>>>(
        attn_bf, Wo_bf, bo, nullptr, nullptr, nullptr, nullptr, out);
}
